// NonLocalBlock_3418793967649
// MI455X (gfx1250) — compile-verified, matrix operand order checked by reading, not run on hardware
//
#include <hip/hip_runtime.h>
#include <hip/hip_bf16.h>
#include <stdint.h>

// ---------------------------------------------------------------------------
// NonLocalBlock for MI455X (gfx1250): bf16 WMMA GEMMs + fp32 softmax/accum.
// B=16, C=512, H=W=32 (N=1024 tokens), 32 groups.
// Round 4: native fp32->bf16 converts (v_cvt_bf16 path) instead of integer
//          emulation; near-scope prefetch. Async staging + WMMA unchanged.
// ---------------------------------------------------------------------------

#define B_ 16
#define C_ 512
#define N_ 1024

typedef __bf16 v16bf __attribute__((ext_vector_type(16)));
typedef float  v8f   __attribute__((ext_vector_type(8)));
typedef unsigned int uint4v __attribute__((ext_vector_type(4)));
typedef int    int4v __attribute__((ext_vector_type(4)));

union FragB { v16bf v; uint4v q[2]; };

// ---- CDNA5 async global->LDS copy (bypasses VGPRs, tracked by ASYNCcnt) ----
#if __has_builtin(__builtin_amdgcn_global_load_async_to_lds_b128)
#define HAS_ASYNC_LDS 1
typedef __attribute__((address_space(1))) int4v* gi4p_t;   // global src
typedef __attribute__((address_space(3))) int4v* li4p_t;   // LDS dst
#else
#define HAS_ASYNC_LDS 0
#endif

__device__ __forceinline__ void cp_lds_b128(unsigned short* l, const unsigned short* g) {
#if HAS_ASYNC_LDS
    __builtin_amdgcn_global_load_async_to_lds_b128((gi4p_t)g, (li4p_t)l, 0, 0);
#else
    *(uint4v*)l = *(const uint4v*)g;
#endif
}

__device__ __forceinline__ void cp_lds_b256(unsigned short* l, const unsigned short* g) {
    cp_lds_b128(l, g);
    cp_lds_b128(l + 8, g + 8);
}

__device__ __forceinline__ void wait_async() {
#if HAS_ASYNC_LDS
#if __has_builtin(__builtin_amdgcn_s_wait_asynccnt)
    __builtin_amdgcn_s_wait_asynccnt(0);
#else
    asm volatile("s_wait_asynccnt 0" ::: "memory");
#endif
#endif
}

// Native fp32 -> bf16 (RNE fptrunc; lowers to v_cvt_bf16 / v_cvt_pk_bf16).
__device__ __forceinline__ unsigned short f2bf(float f) {
    union { __bf16 b; unsigned short u; } cv;
    cv.b = (__bf16)f;
    return cv.u;
}

__device__ __forceinline__ unsigned pack_bf2(float lo, float hi) {
    typedef __bf16 bf2 __attribute__((ext_vector_type(2)));
    union { bf2 b; unsigned u; } cv;
    cv.b = bf2{(__bf16)lo, (__bf16)hi};
    return cv.u;
}

#define ZERO8F {0.f,0.f,0.f,0.f,0.f,0.f,0.f,0.f}

// Compute 2x4 WMMA tiles (wave tile 32x64) from staged LDS tiles.
// ldsA: [128 rows][32 k] bf16 (row-major, pitch 32)
// ldsB: [128 cols][32 k] bf16 (col holds B^T row => B fragment layout == A's)
__device__ __forceinline__ void wmma_step(v8f acc[2][4],
                                          const unsigned short* ldsA,
                                          const unsigned short* ldsB,
                                          int lane, int wm, int wn) {
    const int half = lane >> 4;     // K-half select
    const int mr   = lane & 15;     // row/col within 16-subtile
    FragB a[2], bb[4];
#pragma unroll
    for (int i = 0; i < 2; ++i) {
        const unsigned short* p = ldsA + (wm * 32 + i * 16 + mr) * 32;
        a[i].q[0] = *(const uint4v*)(p + 8 * half);        // K = 8h .. 8h+7
        a[i].q[1] = *(const uint4v*)(p + 16 + 8 * half);   // K = 16+8h .. 16+8h+7
    }
#pragma unroll
    for (int j = 0; j < 4; ++j) {
        const unsigned short* p = ldsB + (wn * 64 + j * 16 + mr) * 32;
        bb[j].q[0] = *(const uint4v*)(p + 8 * half);
        bb[j].q[1] = *(const uint4v*)(p + 16 + 8 * half);
    }
#pragma unroll
    for (int i = 0; i < 2; ++i)
#pragma unroll
        for (int j = 0; j < 4; ++j)
            acc[i][j] = __builtin_amdgcn_wmma_f32_16x16x32_bf16(
                false, a[i].v, false, bb[j].v, (short)0, acc[i][j], false, false);
}

// ---------------------------------------------------------------------------
// K1: GroupNorm statistics. One block per (batch, group) = 512 blocks.
// ---------------------------------------------------------------------------
__global__ void __launch_bounds__(256) k_gn_stats(const float* __restrict__ x,
                                                  float* __restrict__ stats) {
    const int bg = blockIdx.x;
    const float4* p = (const float4*)(x + (size_t)bg * 16 * N_);
    float s = 0.f, ss = 0.f;
#pragma unroll
    for (int i = 0; i < 16; ++i) {
        float4 v = p[threadIdx.x + i * 256];
        s  += v.x + v.y + v.z + v.w;
        ss += v.x * v.x + v.y * v.y + v.z * v.z + v.w * v.w;
    }
    __shared__ float rs[256], rq[256];
    rs[threadIdx.x] = s; rq[threadIdx.x] = ss;
    __syncthreads();
    for (int st = 128; st > 0; st >>= 1) {
        if ((int)threadIdx.x < st) {
            rs[threadIdx.x] += rs[threadIdx.x + st];
            rq[threadIdx.x] += rq[threadIdx.x + st];
        }
        __syncthreads();
    }
    if (threadIdx.x == 0) {
        float m   = rs[0] * (1.f / 16384.f);
        float var = rq[0] * (1.f / 16384.f) - m * m;
        stats[bg]       = m;
        stats[512 + bg] = rsqrtf(var + 1e-6f);
    }
}

// ---------------------------------------------------------------------------
// K2: fused GroupNorm-apply + QKV GEMM. Writes Q/K/V token-major bf16 and
//     additionally V^T ([B,C,N] bf16) for fast AV staging.
// Grid: (1536/128, 1024/128, B).
// ---------------------------------------------------------------------------
__global__ void __launch_bounds__(256) k_gn_qkv(
        const float* __restrict__ x, const float* __restrict__ gamma,
        const float* __restrict__ beta, const float* __restrict__ wqkv,
        const float* __restrict__ bqkv, const float* __restrict__ stats,
        unsigned short* __restrict__ qb, unsigned short* __restrict__ kb,
        unsigned short* __restrict__ vb, unsigned short* __restrict__ vbT) {
    __shared__ __align__(16) unsigned short ldsA[128 * 32];
    __shared__ __align__(16) unsigned short ldsB[128 * 32];
    const int b   = blockIdx.z;
    const int m0  = blockIdx.y * 128;   // token rows
    const int cc0 = blockIdx.x * 128;   // output columns in [0,1536)
    const int t = threadIdx.x, lane = t & 31, wave = t >> 5;
    const int wm = wave >> 1, wn = wave & 1;

    v8f acc[2][4];
#pragma unroll
    for (int i = 0; i < 2; ++i)
#pragma unroll
        for (int j = 0; j < 4; ++j) acc[i][j] = (v8f)ZERO8F;

    for (int k0 = 0; k0 < C_; k0 += 32) {
        // stage A: normalize x[b, c, n] on the fly -> ldsA[n_local][c_local] bf16
        {
            const int cl = t >> 3, nsub = t & 7;
            const int c  = k0 + cl;
            const int g  = c >> 4;
            const float mean = stats[b * 32 + g];
            const float rstd = stats[512 + b * 32 + g];
            const float sc = gamma[c] * rstd;
            const float sh = beta[c] - mean * sc;
            const float4* src = (const float4*)(x + (size_t)(b * C_ + c) * N_ + m0 + nsub * 16);
            if (k0 + 32 < C_)   // prefetch next k-tile's x (same n, c+32)
                __builtin_prefetch((const char*)src + (size_t)32 * N_ * 4, 0, 3);
            unsigned short* dst = ldsA + cl;
#pragma unroll
            for (int v4i = 0; v4i < 4; ++v4i) {
                float4 v = src[v4i];
                int nl = nsub * 16 + v4i * 4;
                dst[(nl + 0) * 32] = f2bf(v.x * sc + sh);
                dst[(nl + 1) * 32] = f2bf(v.y * sc + sh);
                dst[(nl + 2) * 32] = f2bf(v.z * sc + sh);
                dst[(nl + 3) * 32] = f2bf(v.w * sc + sh);
            }
        }
        // stage B: ldsB[c_out_local][c_in_local] = bf16(wqkv[cc0+row, k0+..])
        {
            const int row = t >> 1, half = t & 1;
            const float4* src = (const float4*)(wqkv + (size_t)(cc0 + row) * C_ + k0 + half * 16);
            unsigned int* dst = (unsigned int*)(ldsB + row * 32 + half * 16);
#pragma unroll
            for (int v4i = 0; v4i < 4; ++v4i) {
                float4 v = src[v4i];
                dst[v4i * 2 + 0] = pack_bf2(v.x, v.y);
                dst[v4i * 2 + 1] = pack_bf2(v.z, v.w);
            }
        }
        __syncthreads();
        wmma_step(acc, ldsA, ldsB, lane, wm, wn);
        __syncthreads();
    }
    // epilogue: +bias, route to Q/K/V (and V^T), bf16
    {
        const int half = lane >> 4, col16 = lane & 15;
#pragma unroll
        for (int i = 0; i < 2; ++i)
#pragma unroll
            for (int j = 0; j < 4; ++j) {
                const int cc = cc0 + wn * 64 + j * 16 + col16;
                const float bias = bqkv[cc];
                unsigned short* dstbuf = (cc < 512) ? qb : (cc < 1024) ? kb : vb;
                const int cw = cc & 511;
#pragma unroll
                for (int r = 0; r < 8; ++r) {
                    const int row = m0 + wm * 32 + i * 16 + r + 8 * half;
                    const unsigned short bfv = f2bf(acc[i][j][r] + bias);
                    dstbuf[((size_t)b * N_ + row) * C_ + cw] = bfv;
                    if (cc >= 1024)
                        vbT[((size_t)b * C_ + cw) * N_ + row] = bfv;
                }
            }
    }
}

// ---------------------------------------------------------------------------
// K3: S = (Q K^T) * 1/sqrt(C). Double-buffered async staging.
// Grid: (1024/128 keys, 1024/128 queries, B).
// ---------------------------------------------------------------------------
__global__ void __launch_bounds__(256) k_qk(const unsigned short* __restrict__ qb,
                                            const unsigned short* __restrict__ kb,
                                            float* __restrict__ sbuf) {
    __shared__ __align__(16) unsigned short ldsA[2][128 * 32];
    __shared__ __align__(16) unsigned short ldsB[2][128 * 32];
    const int b = blockIdx.z, q0 = blockIdx.y * 128, key0 = blockIdx.x * 128;
    const int t = threadIdx.x, lane = t & 31, wave = t >> 5;
    const int wm = wave >> 1, wn = wave & 1;
    const int row = t >> 1, half = t & 1;

    const unsigned short* gA = qb + ((size_t)b * N_ + q0 + row) * C_ + half * 16;
    const unsigned short* gB = kb + ((size_t)b * N_ + key0 + row) * C_ + half * 16;
    const int loff = row * 32 + half * 16;

    v8f acc[2][4];
#pragma unroll
    for (int i = 0; i < 2; ++i)
#pragma unroll
        for (int j = 0; j < 4; ++j) acc[i][j] = (v8f)ZERO8F;

    cp_lds_b256(ldsA[0] + loff, gA);
    cp_lds_b256(ldsB[0] + loff, gB);
    wait_async();
    __syncthreads();
    int cur = 0;
    for (int k0 = 0; k0 < C_; k0 += 32) {
        if (k0 + 32 < C_) {
            cp_lds_b256(ldsA[cur ^ 1] + loff, gA + k0 + 32);
            cp_lds_b256(ldsB[cur ^ 1] + loff, gB + k0 + 32);
        }
        wmma_step(acc, ldsA[cur], ldsB[cur], lane, wm, wn);
        wait_async();
        __syncthreads();
        cur ^= 1;
    }

    const float scale = 0.04419417382415922f;   // 1/sqrt(512)
    const int h2 = lane >> 4, col16 = lane & 15;
#pragma unroll
    for (int i = 0; i < 2; ++i)
#pragma unroll
        for (int j = 0; j < 4; ++j) {
            const int key = key0 + wn * 64 + j * 16 + col16;
#pragma unroll
            for (int r = 0; r < 8; ++r) {
                const int qr = q0 + wm * 32 + i * 16 + r + 8 * h2;
                sbuf[((size_t)b * N_ + qr) * N_ + key] = acc[i][j][r] * scale;
            }
        }
}

// ---------------------------------------------------------------------------
// K4: row softmax; writes P as bf16 in-place over first half of each S row.
// ---------------------------------------------------------------------------
__global__ void __launch_bounds__(256) k_softmax(float* __restrict__ sbuf) {
    const int q = blockIdx.x, b = blockIdx.y, t = threadIdx.x;
    float* row = sbuf + ((size_t)b * N_ + q) * N_;
    float v[4];
#pragma unroll
    for (int k = 0; k < 4; ++k) v[k] = row[t + k * 256];

    __shared__ float red[256];
    float m = fmaxf(fmaxf(v[0], v[1]), fmaxf(v[2], v[3]));
    red[t] = m; __syncthreads();
    for (int st = 128; st > 0; st >>= 1) {
        if (t < st) red[t] = fmaxf(red[t], red[t + st]);
        __syncthreads();
    }
    const float M = red[0];
    __syncthreads();

    float e[4], s = 0.f;
#pragma unroll
    for (int k = 0; k < 4; ++k) { e[k] = __expf(v[k] - M); s += e[k]; }
    red[t] = s; __syncthreads();
    for (int st = 128; st > 0; st >>= 1) {
        if (t < st) red[t] += red[t + st];
        __syncthreads();
    }
    const float inv = 1.f / red[0];
    __syncthreads();

    unsigned short* prow = (unsigned short*)row;   // bf16 P in first 2KB of row
#pragma unroll
    for (int k = 0; k < 4; ++k) prow[t + k * 256] = f2bf(e[k] * inv);
}

// ---------------------------------------------------------------------------
// K5: Y = P V, staged from P rows and V^T; double-buffered async.
// Grid: (C/128, N/128, B). K-dim = keys (1024).
// ---------------------------------------------------------------------------
__global__ void __launch_bounds__(256) k_av(const float* __restrict__ sbuf,
                                            const unsigned short* __restrict__ vbT,
                                            unsigned short* __restrict__ yb) {
    __shared__ __align__(16) unsigned short ldsA[2][128 * 32];
    __shared__ __align__(16) unsigned short ldsB[2][128 * 32];
    const int b = blockIdx.z, q0 = blockIdx.y * 128, c0 = blockIdx.x * 128;
    const int t = threadIdx.x, lane = t & 31, wave = t >> 5;
    const int wm = wave >> 1, wn = wave & 1;
    const int row = t >> 1, half = t & 1;

    // A: P row (bf16 packed in fp32-pitch rows)
    const unsigned short* gA =
        (const unsigned short*)(sbuf + ((size_t)b * N_ + q0 + row) * N_) + half * 16;
    // B: V^T row c -> contiguous along keys
    const unsigned short* gB = vbT + ((size_t)b * C_ + c0 + row) * N_ + half * 16;
    const int loff = row * 32 + half * 16;

    v8f acc[2][4];
#pragma unroll
    for (int i = 0; i < 2; ++i)
#pragma unroll
        for (int j = 0; j < 4; ++j) acc[i][j] = (v8f)ZERO8F;

    cp_lds_b256(ldsA[0] + loff, gA);
    cp_lds_b256(ldsB[0] + loff, gB);
    wait_async();
    __syncthreads();
    int cur = 0;
    for (int k0 = 0; k0 < N_; k0 += 32) {
        if (k0 + 32 < N_) {
            cp_lds_b256(ldsA[cur ^ 1] + loff, gA + k0 + 32);
            cp_lds_b256(ldsB[cur ^ 1] + loff, gB + k0 + 32);
        }
        wmma_step(acc, ldsA[cur], ldsB[cur], lane, wm, wn);
        wait_async();
        __syncthreads();
        cur ^= 1;
    }

    const int h2 = lane >> 4, col16 = lane & 15;
#pragma unroll
    for (int i = 0; i < 2; ++i)
#pragma unroll
        for (int j = 0; j < 4; ++j) {
            const int c = c0 + wn * 64 + j * 16 + col16;
#pragma unroll
            for (int r = 0; r < 8; ++r) {
                const int qr = q0 + wm * 32 + i * 16 + r + 8 * h2;
                yb[((size_t)b * N_ + qr) * C_ + c] = f2bf(acc[i][j][r]);
            }
        }
}

// ---------------------------------------------------------------------------
// K6: out = Y Wproj^T + bias + x (residual), fp32 NCHW output.
// A staged async (double-buffered); B converted fp32->bf16 in VGPRs.
// Grid: (C/128, N/128, B).
// ---------------------------------------------------------------------------
__global__ void __launch_bounds__(256) k_proj(const unsigned short* __restrict__ yb,
                                              const float* __restrict__ wproj,
                                              const float* __restrict__ bproj,
                                              const float* __restrict__ x,
                                              float* __restrict__ out) {
    __shared__ __align__(16) unsigned short ldsA[2][128 * 32];
    __shared__ __align__(16) unsigned short ldsB[2][128 * 32];
    const int b = blockIdx.z, q0 = blockIdx.y * 128, c0 = blockIdx.x * 128;
    const int t = threadIdx.x, lane = t & 31, wave = t >> 5;
    const int wm = wave >> 1, wn = wave & 1;
    const int row = t >> 1, half = t & 1;

    const unsigned short* gA = yb + ((size_t)b * N_ + q0 + row) * C_ + half * 16;
    const float* gW = wproj + (size_t)(c0 + row) * C_ + half * 16;
    const int loff = row * 32 + half * 16;

    v8f acc[2][4];
#pragma unroll
    for (int i = 0; i < 2; ++i)
#pragma unroll
        for (int j = 0; j < 4; ++j) acc[i][j] = (v8f)ZERO8F;

    auto stageB = [&](int buf, int k0) {
        const float4* src = (const float4*)(gW + k0);
        unsigned int* dst = (unsigned int*)(ldsB[buf] + loff);
#pragma unroll
        for (int v4i = 0; v4i < 4; ++v4i) {
            float4 v = src[v4i];
            dst[v4i * 2 + 0] = pack_bf2(v.x, v.y);
            dst[v4i * 2 + 1] = pack_bf2(v.z, v.w);
        }
    };

    cp_lds_b256(ldsA[0] + loff, gA);
    stageB(0, 0);
    wait_async();
    __syncthreads();
    int cur = 0;
    for (int k0 = 0; k0 < C_; k0 += 32) {
        if (k0 + 32 < C_) {
            cp_lds_b256(ldsA[cur ^ 1] + loff, gA + k0 + 32);
            stageB(cur ^ 1, k0 + 32);
        }
        wmma_step(acc, ldsA[cur], ldsB[cur], lane, wm, wn);
        wait_async();
        __syncthreads();
        cur ^= 1;
    }

    const int h2 = lane >> 4, col16 = lane & 15;
#pragma unroll
    for (int i = 0; i < 2; ++i)
#pragma unroll
        for (int j = 0; j < 4; ++j) {
            const int cout = c0 + wn * 64 + j * 16 + col16;
            const float bias = bproj[cout];
#pragma unroll
            for (int r = 0; r < 8; ++r) {
                const int n = q0 + wm * 32 + i * 16 + r + 8 * h2;
                const size_t oi = ((size_t)b * C_ + cout) * N_ + n;
                out[oi] = acc[i][j][r] + bias + x[oi];
            }
        }
}

// ---------------------------------------------------------------------------
extern "C" void kernel_launch(void* const* d_in, const int* in_sizes, int n_in,
                              void* d_out, int out_size, void* d_ws, size_t ws_size,
                              hipStream_t stream) {
    (void)in_sizes; (void)n_in; (void)out_size; (void)ws_size;
    const float* x     = (const float*)d_in[0];
    const float* gamma = (const float*)d_in[1];
    const float* beta  = (const float*)d_in[2];
    const float* wqkv  = (const float*)d_in[3];
    const float* bqkv  = (const float*)d_in[4];
    const float* wproj = (const float*)d_in[5];
    const float* bproj = (const float*)d_in[6];
    float* out = (float*)d_out;

    // ws: stats(4KB) | Q | K | V | S(64MB fp32 / P bf16) | Y | V^T   (bf16 = 16MB each)
    float* stats = (float*)d_ws;
    unsigned short* qb  = (unsigned short*)((char*)d_ws + 4096);
    unsigned short* kb  = qb + (size_t)B_ * N_ * C_;
    unsigned short* vb  = kb + (size_t)B_ * N_ * C_;
    float* sbuf = (float*)(vb + (size_t)B_ * N_ * C_);
    unsigned short* yb  = (unsigned short*)(sbuf + (size_t)B_ * N_ * N_);
    unsigned short* vbT = yb + (size_t)B_ * N_ * C_;

    k_gn_stats<<<dim3(B_ * 32), 256, 0, stream>>>(x, stats);
    k_gn_qkv <<<dim3(12, 8, B_), 256, 0, stream>>>(x, gamma, beta, wqkv, bqkv, stats,
                                                   qb, kb, vb, vbT);
    k_qk     <<<dim3(8, 8, B_), 256, 0, stream>>>(qb, kb, sbuf);
    k_softmax<<<dim3(N_, B_), 256, 0, stream>>>(sbuf);
    k_av     <<<dim3(4, 8, B_), 256, 0, stream>>>(sbuf, vbT, yb);
    k_proj   <<<dim3(4, 8, B_), 256, 0, stream>>>(yb, wproj, bproj, x, out);
}
